// NeuralCDEDecoder_68908455297337
// MI455X (gfx1250) — matrix operand order repaired and checked by reading, not compiled, not run on hardware
//
#include <hip/hip_runtime.h>

// ---------------------------------------------------------------------------
// Neural CDE + decoder, persistent-workgroup WMMA implementation for gfx1250.
//
// B=512, T=257, IN=32, H=128, BN=256, OUT=64, K_SUB=4 (RK4) =>
// 256*4*4 = 4096 sequential vector-field evals; batch-parallel otherwise.
// grid = 32 workgroups x 256 threads (8 wave32); each workgroup owns 16
// batch rows (one WMMA M tile) for the whole integration. Activations and
// RK state live in LDS; fp16-packed weights stream from L2 (2.2 MB total,
// trivially resident in the 192 MB L2).
//
// v2 changes vs v1:
//  - A-operand fragments hoisted out of the N-tile loops (phase 3: 16x fewer
//    ds_load_b128; A for all 32 tiles is the same h2 block).
//  - phase 3 processes the two parity tiles of each h together: two
//    independent WMMA chains, parities summed before the lane butterfly
//    (half the shuffles), vf written once -> no zero pass, no LDS RMW.
//  - b3 staged to LDS once (kills 32 global bias loads per wave per eval).
// ---------------------------------------------------------------------------

typedef __attribute__((ext_vector_type(16))) _Float16 v16h;
typedef __attribute__((ext_vector_type(8)))  _Float16 v8h;
typedef __attribute__((ext_vector_type(8)))  float    v8f;

#define NB     512
#define NT_T   257
#define NIV    256      // T-1 intervals
#define NIN    32
#define NH     128
#define NBN    256
#define NOUT   64
#define ROWS   16       // batch rows per workgroup (one WMMA M tile)

// ---- weight packing into WMMA B-operand layout -----------------------------
// B operand of v_wmma_f32_16x16x32_f16 is 32(K) x 16(N) fp16 per tile:
//   lane l holds column n = l&15; lanes 0-15 hold K 0..7,16..23, lanes 16-31
//   hold K 8..15,24..31 (matching load_A). Tiles stored (kt,nt) tile-major
//   so each lane reads 32 contiguous bytes per K-step.
__global__ void pack_weight(const float* __restrict__ src, _Float16* __restrict__ dst,
                            int K, int N) {
  int idx = blockIdx.x * 256 + threadIdx.x;
  if (idx >= K * N) return;
  int j    = idx & 15;
  int lane = (idx >> 4) & 31;
  int tile = idx >> 9;
  int NT   = N >> 4;
  int nt   = tile % NT;
  int kt   = tile / NT;
  int k = kt * 32 + ((j >> 3) << 4) + ((lane >> 4) << 3) + (j & 7);
  int n = nt * 16 + (lane & 15);
  dst[idx] = (_Float16)src[(size_t)k * N + n];
}

// ---- A-operand load from row-major fp16 LDS --------------------------------
// A operand is 16(M) x 32(K): lane l -> row m = l&15; lanes 0-15 take K 0..7
// and 16..23, lanes 16-31 take K 8..15 and 24..31  => two 16-byte chunks.
__device__ inline v16h load_A(const _Float16* base, int strideH, int k0, int lane) {
  const _Float16* p = base + (size_t)(lane & 15) * strideH + k0 + ((lane >> 4) << 3);
  v8h lo = *(const v8h*)(p);
  v8h hi = *(const v8h*)(p + 16);
  return __builtin_shufflevector(lo, hi, 0,1,2,3,4,5,6,7,8,9,10,11,12,13,14,15);
}

template<int KT>
__device__ inline void load_A_all(const _Float16* base, int strideH, int lane,
                                  v16h* afr) {
#pragma unroll
  for (int kt = 0; kt < KT; ++kt) afr[kt] = load_A(base, strideH, kt * 32, lane);
}

template<int KT, int NT>
__device__ inline v8f gemm_preA(const v16h* afr, const _Float16* __restrict__ Bpk,
                                int nt, int lane) {
  v8f acc = {};
#pragma unroll
  for (int kt = 0; kt < KT; ++kt) {
    v16h b = *(const v16h*)(Bpk + ((size_t)(kt * NT + nt) * 32 + lane) * 16);
    acc = __builtin_amdgcn_wmma_f32_16x16x32_f16(
        false, afr[kt], false, b, (short)0, acc, false, false);
  }
  return acc;
}

// ---- persistent integration kernel -----------------------------------------
__global__ __launch_bounds__(256)
void ncde_persistent(const float* __restrict__ t_g,
                     const float* __restrict__ z0_g,
                     const float* __restrict__ X_g,
                     const float* __restrict__ b1_g,
                     const float* __restrict__ b2_g,
                     const float* __restrict__ b3_g,
                     const float* __restrict__ mb_g,
                     const float* __restrict__ sb_g,
                     const _Float16* __restrict__ W1pk,
                     const _Float16* __restrict__ W2pk,
                     const _Float16* __restrict__ W3pk,
                     const _Float16* __restrict__ mWpk,
                     const _Float16* __restrict__ sWpk,
                     float* __restrict__ out_mean,
                     float* __restrict__ out_std) {
  __shared__ __align__(32) float    zS[ROWS][NH];     // fp32 RK state
  __shared__ __align__(32) float    ksumS[ROWS][NH];  // RK k accumulator
  __shared__ __align__(32) _Float16 zh[ROWS][NH];     // fp16 stage input (A)
  __shared__ __align__(32) _Float16 h1S[ROWS][NBN];
  __shared__ __align__(32) _Float16 h2S[ROWS][NBN];
  __shared__ __align__(32) float    vfS[ROWS][NH];    // vector-field output
  __shared__ __align__(32) float    dxS[ROWS][NIN];   // dX/dt this interval
  __shared__ __align__(32) float    b3L[NIN * NH];    // b3 staged once (16 KB)

  const int tid  = threadIdx.x;
  const int lane = tid & 31;
  const int w    = tid >> 5;            // wave id 0..7
  const int b0   = blockIdx.x * ROWS;   // global batch row base
  const int rb   = (lane >> 4) * 8;     // C-layout row base for this lane half
  const int nl   = lane & 15;

  // stage b3 into LDS; init z from z0
  for (int e = tid; e < NIN * NH; e += 256) b3L[e] = b3_g[e];
  for (int e = tid; e < ROWS * NH; e += 256) {
    int b = e >> 7, h = e & 127;
    float z = z0_g[(size_t)(b0 + b) * NH + h];
    zS[b][h] = z;
    zh[b][h] = (_Float16)z;
  }
  __syncthreads();

  for (int iv = 0; iv < NIV; ++iv) {
    const float dt    = t_g[iv + 1] - t_g[iv];
    const float hstep = dt * 0.25f;     // dt / K_SUB

    // dX/dt for this interval (constant per interval)
    for (int e = tid; e < ROWS * NIN; e += 256) {
      int b = e >> 5, i = e & 31;
      size_t base = ((size_t)(b0 + b) * NT_T + iv) * NIN + i;
      dxS[b][i] = (X_g[base + NIN] - X_g[base]) / dt;
    }
    __syncthreads();

    // per-lane dX/dt registers for the phase-3 contraction:
    // element (VGPR r, lane) of an o-tile has row rb+r and i = parity*16+nl.
    float dx0[8], dx1[8];
#pragma unroll
    for (int r = 0; r < 8; ++r) {
      dx0[r] = dxS[rb + r][nl];
      dx1[r] = dxS[rb + r][16 + nl];
    }

    for (int sub = 0; sub < 4; ++sub) {
#pragma unroll 1
      for (int stage = 0; stage < 4; ++stage) {
        // ---- phase 1: h1 = relu(z @ W1 + b1)   [16,128]x[128,256] ----
        {
          v16h afr[4];
          load_A_all<4>(&zh[0][0], NH, lane, afr);
#pragma unroll
          for (int q = 0; q < 2; ++q) {
            int nt = w + q * 8;
            v8f acc = gemm_preA<4, 16>(afr, W1pk, nt, lane);
            int n = nt * 16 + nl;
            float bias = b1_g[n];
#pragma unroll
            for (int r = 0; r < 8; ++r)
              h1S[rb + r][n] = (_Float16)fmaxf(acc[r] + bias, 0.f);
          }
        }
        __syncthreads();

        // ---- phase 2: h2 = relu(h1 @ W2 + b2)  [16,256]x[256,256] ----
        {
          v16h afr[8];
          load_A_all<8>(&h1S[0][0], NBN, lane, afr);
#pragma unroll
          for (int q = 0; q < 2; ++q) {
            int nt = w + q * 8;
            v8f acc = gemm_preA<8, 16>(afr, W2pk, nt, lane);
            int n = nt * 16 + nl;
            float bias = b2_g[n];
#pragma unroll
            for (int r = 0; r < 8; ++r)
              h2S[rb + r][n] = (_Float16)fmaxf(acc[r] + bias, 0.f);
          }
        }
        __syncthreads();

        // ---- phase 3: o = tanh(h2 @ W3 + b3); vf = einsum(o, dxdt) ----
        // wave w exclusively owns h in [w*16, w*16+16); the two parity tiles
        // (i 0..15 / 16..31) of each h run as two independent WMMA chains and
        // are combined before the lane reduction, so vf is written exactly
        // once -> no zeroing, no read-modify-write.
        {
          v16h afr[8];
          load_A_all<8>(&h2S[0][0], NBN, lane, afr);
#pragma unroll 1
          for (int tt = 0; tt < 32; tt += 2) {
            int nt0 = w * 32 + tt;
            if (tt + 2 < 32)                  // prefetch next pair (L2 hint)
              __builtin_prefetch(W3pk + ((size_t)(nt0 + 2) * 32 + lane) * 16, 0, 1);
            v8f acc0 = {}, acc1 = {};
#pragma unroll
            for (int kt = 0; kt < 8; ++kt) {
              const _Float16* bb =
                  W3pk + ((size_t)(kt * 256 + nt0) * 32 + lane) * 16;
              v16h bv0 = *(const v16h*)(bb);
              v16h bv1 = *(const v16h*)(bb + 512);   // nt0+1 tile
              acc0 = __builtin_amdgcn_wmma_f32_16x16x32_f16(
                  false, afr[kt], false, bv0, (short)0, acc0, false, false);
              acc1 = __builtin_amdgcn_wmma_f32_16x16x32_f16(
                  false, afr[kt], false, bv1, (short)0, acc1, false, false);
            }
            int n0 = nt0 * 16 + nl;
            float bias0 = b3L[n0];
            float bias1 = b3L[n0 + 16];
            int hcol = (nt0 >> 1);               // = w*16 + tt/2
#pragma unroll
            for (int r = 0; r < 8; ++r) {
              float val = tanhf(acc0[r] + bias0) * dx0[r]
                        + tanhf(acc1[r] + bias1) * dx1[r];
              // reduce the 16 columns held across each 16-lane half
              val += __shfl_xor(val, 1, 32);
              val += __shfl_xor(val, 2, 32);
              val += __shfl_xor(val, 4, 32);
              val += __shfl_xor(val, 8, 32);
              if (nl == 0) vfS[rb + r][hcol] = val;   // exclusive, write-once
            }
          }
        }
        __syncthreads();

        // ---- RK4 state update (elementwise, fp32 in LDS) ----
        for (int e = tid; e < ROWS * NH; e += 256) {
          int b = e >> 7, h = e & 127;
          float k = vfS[b][h];
          float z = zS[b][h];
          if (stage == 0) {
            ksumS[b][h] = k;
            zh[b][h] = (_Float16)(z + 0.5f * hstep * k);
          } else if (stage == 1) {
            ksumS[b][h] += 2.f * k;
            zh[b][h] = (_Float16)(z + 0.5f * hstep * k);
          } else if (stage == 2) {
            ksumS[b][h] += 2.f * k;
            zh[b][h] = (_Float16)(z + hstep * k);
          } else {
            float zn = z + (hstep * (1.f / 6.f)) * (ksumS[b][h] + k);
            zS[b][h] = zn;
            zh[b][h] = (_Float16)zn;   // z(t) for next substep / projection
          }
        }
        __syncthreads();
      } // stage
    }   // sub

    // ---- decoder at t[iv+1]: mean = z@mW+mb ; std = softplus(z@sW+sb) ----
    {
      v16h afr[4];
      load_A_all<4>(&zh[0][0], NH, lane, afr);
      int m  = w >> 2;                   // 0 = mean, 1 = std
      int nt = w & 3;                    // OUT=64 -> 4 N tiles
      const _Float16* Wp = m ? sWpk : mWpk;
      v8f acc = gemm_preA<4, 4>(afr, Wp, nt, lane);
      int n = nt * 16 + nl;
      float bias = m ? sb_g[n] : mb_g[n];
      float* outp = m ? out_std : out_mean;
#pragma unroll
      for (int r = 0; r < 8; ++r) {
        float v = acc[r] + bias;
        if (m) v = (v > 20.f) ? v : log1pf(expf(v));
        outp[((size_t)(b0 + rb + r) * NIV + iv) * NOUT + n] = v;
      }
    }
    // next interval's dxS rewrite is separated from all readers by the
    // __syncthreads immediately after it; zh is stable until next stage-0.
  }
}

// ---------------------------------------------------------------------------
extern "C" void kernel_launch(void* const* d_in, const int* in_sizes, int n_in,
                              void* d_out, int out_size, void* d_ws, size_t ws_size,
                              hipStream_t stream) {
  const float* t  = (const float*)d_in[0];
  const float* z0 = (const float*)d_in[1];
  const float* X  = (const float*)d_in[2];
  const float* W1 = (const float*)d_in[3];
  const float* b1 = (const float*)d_in[4];
  const float* W2 = (const float*)d_in[5];
  const float* b2 = (const float*)d_in[6];
  const float* W3 = (const float*)d_in[7];
  const float* b3 = (const float*)d_in[8];
  const float* mW = (const float*)d_in[9];
  const float* mb = (const float*)d_in[10];
  const float* sW = (const float*)d_in[11];
  const float* sb = (const float*)d_in[12];

  // fp16 packed weights in workspace (all offsets 32B-aligned)
  _Float16* ws   = (_Float16*)d_ws;
  _Float16* W1pk = ws;                       // 128*256   = 32768 halves
  _Float16* W2pk = ws + 32768;               // 256*256   = 65536
  _Float16* W3pk = ws + 98304;               // 256*4096  = 1048576
  _Float16* mWpk = ws + 1146880;             // 128*64    = 8192
  _Float16* sWpk = ws + 1155072;             // 128*64    = 8192

  pack_weight<<<(128 * 256  + 255) / 256, 256, 0, stream>>>(W1, W1pk, 128, 256);
  pack_weight<<<(256 * 256  + 255) / 256, 256, 0, stream>>>(W2, W2pk, 256, 256);
  pack_weight<<<(256 * 4096 + 255) / 256, 256, 0, stream>>>(W3, W3pk, 256, 4096);
  pack_weight<<<(128 * 64   + 255) / 256, 256, 0, stream>>>(mW, mWpk, 128, 64);
  pack_weight<<<(128 * 64   + 255) / 256, 256, 0, stream>>>(sW, sWpk, 128, 64);

  float* out_mean = (float*)d_out;
  float* out_std  = out_mean + (size_t)NB * NIV * NOUT;  // 8388608

  ncde_persistent<<<NB / ROWS, 256, 0, stream>>>(
      t, z0, X, b1, b2, b3, mb, sb, W1pk, W2pk, W3pk, mWpk, sWpk,
      out_mean, out_std);
}
